// Block_1700807050120
// MI455X (gfx1250) — matrix operand order repaired and checked by reading, not compiled, not run on hardware
//
#include <hip/hip_runtime.h>
#include <hip/hip_bf16.h>
#include <math.h>

typedef __attribute__((ext_vector_type(16))) _Float16 v16h;
typedef __attribute__((ext_vector_type(8)))  float    v8f;

__device__ __forceinline__ v8f wmma_f16(v16h a, v16h b, v8f c) {
  // D = A(16x32 f16) * B(32x16 f16) + C(16x16 f32)
  return __builtin_amdgcn_wmma_f32_16x16x32_f16(false, a, false, b, (short)0, c, false, false);
}

// Fragment K-index maps (contiguous order, same for A and B):
// A-matrix 16x32 f16: half j -> K = j + 16*hf   (hf = lane>=16)
__device__ __forceinline__ int kA(int j, int hf) { return j + hf * 16; }
// B-matrix 32x16 f16: half j -> K = j + 16*hf
__device__ __forceinline__ int kB(int j, int hf) { return j + hf * 16; }

// ---------------------------------------------------------------------------
// CDNA5 async global->LDS copy (ASYNCcnt), 16 bytes per lane.
// ---------------------------------------------------------------------------
__device__ __forceinline__ unsigned lds_addr(const void* p) {
  // flat shared pointer: low 32 bits are the LDS byte offset
  return (unsigned)(uintptr_t)p;
}
__device__ __forceinline__ void async_copy_b128(const void* gptr, unsigned ldsoff) {
  asm volatile("global_load_async_to_lds_b128 %0, %1, off"
               :: "v"(ldsoff), "v"(gptr) : "memory");
}
__device__ __forceinline__ void wait_async0() {
  asm volatile("s_wait_asynccnt 0x0" ::: "memory");
}

// ---------------------------------------------------------------------------
// Fused f32->f16 convert + transpose: in[K,N] f32 -> out[N,K] f16
// ---------------------------------------------------------------------------
__global__ void __launch_bounds__(256) cvt_transpose_f16(
    const float* __restrict__ in, _Float16* __restrict__ out, int K, int N) {
  __shared__ _Float16 tile[32][33];
  int k0 = blockIdx.y * 32, n0 = blockIdx.x * 32;
  int tx = threadIdx.x & 31, ty = threadIdx.x >> 5;  // 32 x 8
  #pragma unroll
  for (int r = 0; r < 4; ++r)
    tile[ty + r * 8][tx] = (_Float16)in[(size_t)(k0 + ty + r * 8) * N + n0 + tx];
  __syncthreads();
  #pragma unroll
  for (int r = 0; r < 4; ++r)
    out[(size_t)(n0 + ty + r * 8) * K + k0 + tx] = tile[tx][ty + r * 8];
}

// ---------------------------------------------------------------------------
// f16 transpose of the V section of qkv: vT[(b*H+h)*64 + d][key] = V[key][d]
// ---------------------------------------------------------------------------
__global__ void __launch_bounds__(256) transpose_v(
    const _Float16* __restrict__ qkv, _Float16* __restrict__ vT, int N) {
  __shared__ _Float16 tile[32][33];
  int bh = blockIdx.z;                 // b*16 + h
  int b = bh >> 4, h = bh & 15;
  int key0 = blockIdx.x * 32, d0 = blockIdx.y * 32;
  const _Float16* in = qkv + (size_t)(b * N) * 3072 + 2048 + h * 64;
  _Float16* out = vT + (size_t)bh * 64 * N;
  int tx = threadIdx.x & 31, ty = threadIdx.x >> 5;  // 32 x 8
  #pragma unroll
  for (int r = 0; r < 4; ++r)
    tile[ty + r * 8][tx] = in[(size_t)(key0 + ty + r * 8) * 3072 + d0 + tx];
  __syncthreads();
  #pragma unroll
  for (int r = 0; r < 4; ++r)
    out[(size_t)(d0 + ty + r * 8) * N + key0 + tx] = tile[tx][ty + r * 8];
}

// ---------------------------------------------------------------------------
// Row LayerNorm: one block per row of C=1024, f32 in, f16 out
// ---------------------------------------------------------------------------
__global__ void __launch_bounds__(256) layernorm_rows(
    const float* __restrict__ x, const float* __restrict__ g,
    const float* __restrict__ bta, _Float16* __restrict__ out, int C) {
  __shared__ float sh[2][8];
  int row = blockIdx.x;
  const float* xr = x + (size_t)row * C;
  float s = 0.f, ss = 0.f;
  for (int i = threadIdx.x; i < C; i += 256) { float v = xr[i]; s += v; ss += v * v; }
  #pragma unroll
  for (int m = 1; m < 32; m <<= 1) { s += __shfl_xor(s, m, 32); ss += __shfl_xor(ss, m, 32); }
  int wave = threadIdx.x >> 5, lane = threadIdx.x & 31;
  if (lane == 0) { sh[0][wave] = s; sh[1][wave] = ss; }
  __syncthreads();
  if (wave == 0) {
    s  = (lane < 8) ? sh[0][lane] : 0.f;
    ss = (lane < 8) ? sh[1][lane] : 0.f;
    #pragma unroll
    for (int m = 1; m < 8; m <<= 1) { s += __shfl_xor(s, m, 32); ss += __shfl_xor(ss, m, 32); }
    if (lane == 0) { sh[0][0] = s; sh[1][0] = ss; }
  }
  __syncthreads();
  float mu = sh[0][0] / (float)C;
  float var = sh[1][0] / (float)C - mu * mu;
  float rstd = rsqrtf(var + 1e-6f);
  for (int i = threadIdx.x; i < C; i += 256)
    out[(size_t)row * C + i] = (_Float16)((xr[i] - mu) * rstd * g[i] + bta[i]);
}

// ---------------------------------------------------------------------------
// Tiled WMMA GEMM: C[M,N] = A[M,K](f16,row-major) * Bt[N,K](f16, = B^T)
// Block tile 128x128, K-step 32, 8 waves (4x2), each wave 32x64 (2x4 WMMA).
// Async global->LDS staging, double-buffered.
// EPI: 0 = store f16; 1 = +bias +residual -> f32; 2 = +bias +GELU -> f16
// ---------------------------------------------------------------------------
constexpr int BM = 128, BN = 128, BK = 32;
constexpr int LDT = BK + 8;     // 40 halves, 80B row stride (both tiles)

template <int EPI>
__global__ void __launch_bounds__(256) gemm_wmma(
    const _Float16* __restrict__ A, const _Float16* __restrict__ Bt,
    int M, int N, int K,
    const float* __restrict__ bias, const float* __restrict__ resid,
    _Float16* __restrict__ outH, float* __restrict__ outF) {
  __shared__ _Float16 As[2][BM * LDT];
  __shared__ _Float16 Bs[2][BN * LDT];

  int tid = threadIdx.x;
  int lane = tid & 31, wave = tid >> 5;
  int wm = wave >> 1, wn = wave & 1;      // 4 x 2 wave grid -> 32x64 per wave
  int m0 = blockIdx.y * BM, n0 = blockIdx.x * BN;
  int hf = lane >> 4, rr = lane & 15;

  auto stage = [&](int k0, int buf) {
    int arow = tid >> 2;            // 0..63
    int ac = (tid & 3) * 8;         // 0,8,16,24
    #pragma unroll
    for (int rep = 0; rep < 2; ++rep) {
      int r2 = arow + rep * 64;
      async_copy_b128(A + (size_t)(m0 + r2) * K + k0 + ac,
                      lds_addr(&As[buf][r2 * LDT + ac]));
    }
    int brow = tid >> 1;            // 0..127
    int bc = (tid & 1) * 16;        // 0,16
    const _Float16* bg = Bt + (size_t)(n0 + brow) * K + k0 + bc;
    async_copy_b128(bg,     lds_addr(&Bs[buf][brow * LDT + bc]));
    async_copy_b128(bg + 8, lds_addr(&Bs[buf][brow * LDT + bc + 8]));
  };

  v8f acc[2][4] = {};
  int nk = K / BK;
  stage(0, 0);
  for (int ks = 0; ks < nk; ++ks) {
    wait_async0();        // this wave's tile-ks copies have landed in LDS
    __syncthreads();      // everyone's have; all readers of the other buffer done
    if (ks + 1 < nk) stage((ks + 1) * BK, (ks + 1) & 1);   // overlap with math
    int cur = ks & 1;

    v16h af[2], bf[4];
    #pragma unroll
    for (int i = 0; i < 2; ++i) {
      const _Float16* ab = &As[cur][(wm * 32 + i * 16 + rr) * LDT];
      #pragma unroll
      for (int j = 0; j < 16; ++j) af[i][j] = ab[kA(j, hf)];
    }
    #pragma unroll
    for (int t = 0; t < 4; ++t) {
      const _Float16* bb = &Bs[cur][(wn * 64 + t * 16 + rr) * LDT];
      #pragma unroll
      for (int j = 0; j < 16; ++j) bf[t][j] = bb[kB(j, hf)];
    }
    #pragma unroll
    for (int i = 0; i < 2; ++i)
      #pragma unroll
      for (int t = 0; t < 4; ++t)
        acc[i][t] = wmma_f16(af[i], bf[t], acc[i][t]);
  }

  // --- epilogue (C/D layout: row = r + 8*hf, col = rr) ---
  int orow = m0 + wm * 32;
  int ocol = n0 + wn * 64;
  #pragma unroll
  for (int i = 0; i < 2; ++i) {
    #pragma unroll
    for (int t = 0; t < 4; ++t) {
      #pragma unroll
      for (int r = 0; r < 8; ++r) {
        int row = orow + i * 16 + r + hf * 8;
        int col = ocol + t * 16 + rr;
        float v = acc[i][t][r];
        if (EPI == 0) {
          outH[(size_t)row * N + col] = (_Float16)v;
        } else if (EPI == 1) {
          v += bias[col] + resid[(size_t)row * N + col];
          outF[(size_t)row * N + col] = v;
        } else {
          v += bias[col];
          v = 0.5f * v * (1.0f + erff(v * 0.70710678118654752f)); // exact GELU
          outH[(size_t)row * N + col] = (_Float16)v;
        }
      }
    }
  }
}

// ---------------------------------------------------------------------------
// Flash attention. qkv layout: [B*N, 3072] f16 (q | k), vT: [(b*H+h)*64+d][key].
// Block = 256 threads (8 waves), covers 128 query rows for one (b,h).
// K tiles (32 keys x 64 d) and V^T tiles (64 d x 32 keys) async-staged,
// double-buffered; every fragment gather is contiguous ds_load_b128.
// ---------------------------------------------------------------------------
constexpr int LDK = 72;   // K tile row stride (64 + 8 halves)
constexpr int LDV = 40;   // V^T tile row stride (32 + 8 halves)

__global__ void __launch_bounds__(256) attn_flash(
    const _Float16* __restrict__ qkv, const _Float16* __restrict__ vT,
    _Float16* __restrict__ y, int N) {
  constexpr int LDQ = 3072, C = 1024;
  int b = blockIdx.z, h = blockIdx.y;
  int wave = threadIdx.x >> 5, lane = threadIdx.x & 31;
  int hf = lane >> 4, rr = lane & 15;
  int q0 = blockIdx.x * 128 + wave * 16;

  __shared__ _Float16 ktile[2][32 * LDK];   // [key 0..31][d 0..63]
  __shared__ _Float16 vtile[2][64 * LDV];   // [d 0..63][key 0..31]
  __shared__ _Float16 pbuf[8][16 * 36];     // per-wave P transpose staging
  _Float16* pt = pbuf[wave];

  const _Float16* qbase = qkv + ((size_t)(b * N + q0)) * LDQ + h * 64;
  const _Float16* kbase = qkv + ((size_t)(b * N)) * LDQ + 1024 + h * 64;
  const _Float16* vbase = vT + (size_t)(b * 16 + h) * 64 * N;

  auto stageKV = [&](int key0, int buf) {
    // K tile: 32 rows x 64 halves = 256 x 16B chunks, one per thread
    int krow = threadIdx.x >> 3;         // 0..31
    int kc = (threadIdx.x & 7) * 8;      // 0..56
    async_copy_b128(kbase + (size_t)(key0 + krow) * LDQ + kc,
                    lds_addr(&ktile[buf][krow * LDK + kc]));
    // V^T tile: 64 rows x 32 halves = 256 x 16B chunks, one per thread
    int vrow = threadIdx.x >> 2;         // 0..63
    int vc = (threadIdx.x & 3) * 8;      // 0..24
    async_copy_b128(vbase + (size_t)vrow * N + key0 + vc,
                    lds_addr(&vtile[buf][vrow * LDV + vc]));
  };

  // q A-fragments: two K-slices of 32 over Dh=64
  v16h qa[2];
  #pragma unroll
  for (int s = 0; s < 2; ++s)
    #pragma unroll
    for (int j = 0; j < 16; ++j)
      qa[s][j] = qbase[(size_t)rr * LDQ + s * 32 + kA(j, hf)];

  v8f o[4] = {};
  float mrow[8], lrow[8];
  #pragma unroll
  for (int r = 0; r < 8; ++r) { mrow[r] = -1e30f; lrow[r] = 0.f; }

  stageKV(0, 0);
  int it = 0;
  for (int key0 = 0; key0 < N; key0 += 32, ++it) {
    wait_async0();
    __syncthreads();
    if (key0 + 32 < N) stageKV(key0 + 32, (it + 1) & 1);
    int cur = it & 1;
    const _Float16* kt = ktile[cur];
    const _Float16* vt = vtile[cur];

    // S = q @ k^T : two 16x16 tiles (keys 0-15, 16-31), K-loop over Dh
    v8f s0 = {}, s1 = {};
    #pragma unroll
    for (int s = 0; s < 2; ++s) {
      v16h kb0, kb1;
      #pragma unroll
      for (int j = 0; j < 16; ++j) {
        kb0[j] = kt[(rr) * LDK + s * 32 + kB(j, hf)];        // B frag, n=key
        kb1[j] = kt[(16 + rr) * LDK + s * 32 + kB(j, hf)];
      }
      s0 = wmma_f16(qa[s], kb0, s0);
      s1 = wmma_f16(qa[s], kb1, s1);
    }

    // online softmax: row reductions across the 16-lane half via shfl_xor
    const float scale = 0.125f;   // HEAD_DIM^-0.5
    float mx[8];
    #pragma unroll
    for (int r = 0; r < 8; ++r) {
      s0[r] *= scale; s1[r] *= scale;
      mx[r] = fmaxf(s0[r], s1[r]);
    }
    #pragma unroll
    for (int m = 1; m < 16; m <<= 1)
      #pragma unroll
      for (int r = 0; r < 8; ++r) mx[r] = fmaxf(mx[r], __shfl_xor(mx[r], m, 32));

    float sum[8], alpha[8];
    #pragma unroll
    for (int r = 0; r < 8; ++r) {
      float mn = fmaxf(mrow[r], mx[r]);
      alpha[r] = __expf(mrow[r] - mn);
      mrow[r] = mn;
      s0[r] = __expf(s0[r] - mn);
      s1[r] = __expf(s1[r] - mn);
      sum[r] = s0[r] + s1[r];
    }
    #pragma unroll
    for (int m = 1; m < 16; m <<= 1)
      #pragma unroll
      for (int r = 0; r < 8; ++r) sum[r] += __shfl_xor(sum[r], m, 32);
    #pragma unroll
    for (int r = 0; r < 8; ++r) lrow[r] = lrow[r] * alpha[r] + sum[r];
    #pragma unroll
    for (int t = 0; t < 4; ++t)
      #pragma unroll
      for (int r = 0; r < 8; ++r) o[t][r] *= alpha[r];

    // transpose P (C-layout -> row-major) via LDS, reload as A-fragment
    #pragma unroll
    for (int r = 0; r < 8; ++r) {
      int m = r + hf * 8;
      pt[m * 36 + rr]      = (_Float16)s0[r];
      pt[m * 36 + 16 + rr] = (_Float16)s1[r];
    }
    __syncthreads();

    v16h pa;
    #pragma unroll
    for (int j = 0; j < 16; ++j) pa[j] = pt[rr * 36 + kA(j, hf)];

    // O += P @ V: B frag from V^T tile (n=d, k=key) -> contiguous b128 reads
    #pragma unroll
    for (int t = 0; t < 4; ++t) {
      v16h vb;
      #pragma unroll
      for (int j = 0; j < 16; ++j) vb[j] = vt[(t * 16 + rr) * LDV + kB(j, hf)];
      o[t] = wmma_f16(pa, vb, o[t]);
    }
  }

  // finalize: divide by l, write y[b, n, h*64+d] as f16 for proj GEMM
  #pragma unroll
  for (int t = 0; t < 4; ++t)
    #pragma unroll
    for (int r = 0; r < 8; ++r) {
      float v = o[t][r] / lrow[r];
      int m = q0 + r + hf * 8;
      y[(size_t)(b * N + m) * C + h * 64 + t * 16 + rr] = (_Float16)v;
    }
}

// ---------------------------------------------------------------------------
// Host orchestration
// ---------------------------------------------------------------------------
extern "C" void kernel_launch(void* const* d_in, const int* in_sizes, int n_in,
                              void* d_out, int out_size, void* d_ws, size_t ws_size,
                              hipStream_t stream) {
  const float* x      = (const float*)d_in[0];
  const float* ln1_g  = (const float*)d_in[1];
  const float* ln1_b  = (const float*)d_in[2];
  const float* w_qkv  = (const float*)d_in[3];
  const float* w_proj = (const float*)d_in[4];
  const float* b_proj = (const float*)d_in[5];
  const float* ln2_g  = (const float*)d_in[6];
  const float* ln2_b  = (const float*)d_in[7];
  const float* w_mlp1 = (const float*)d_in[8];
  const float* b_mlp1 = (const float*)d_in[9];
  const float* w_mlp2 = (const float*)d_in[10];
  const float* b_mlp2 = (const float*)d_in[11];
  float* out = (float*)d_out;

  constexpr int Bb = 2, Nn = 2048, Cc = 1024, Hh = 16, HID = 4096;
  constexpr int Rows = Bb * Nn;  // 4096

  char* ws = (char*)d_ws;
  size_t off = 0;
  auto alloc = [&](size_t bytes) {
    void* p = ws + off;
    off += (bytes + 255) & ~size_t(255);
    return p;
  };
  _Float16* xn   = (_Float16*)alloc((size_t)Rows * Cc * 2);
  _Float16* qkvb = (_Float16*)alloc((size_t)Rows * 3 * Cc * 2);
  _Float16* vTb  = (_Float16*)alloc((size_t)Bb * Hh * 64 * Nn * 2);
  _Float16* yatt = (_Float16*)alloc((size_t)Rows * Cc * 2);
  float*    x1   = (float*)   alloc((size_t)Rows * Cc * 4);
  _Float16* hbuf = (_Float16*)alloc((size_t)Rows * HID * 2);
  _Float16* wqT  = (_Float16*)alloc((size_t)Cc * 3 * Cc * 2);   // [3C, C]
  _Float16* wpT  = (_Float16*)alloc((size_t)Cc * Cc * 2);       // [C, C]
  _Float16* w1T  = (_Float16*)alloc((size_t)Cc * HID * 2);      // [HID, C]
  _Float16* w2T  = (_Float16*)alloc((size_t)HID * Cc * 2);      // [C, HID]

  // convert + transpose weights to (N,K) f16
  cvt_transpose_f16<<<dim3(3 * Cc / 32, Cc / 32), 256, 0, stream>>>(w_qkv, wqT, Cc, 3 * Cc);
  cvt_transpose_f16<<<dim3(Cc / 32, Cc / 32), 256, 0, stream>>>(w_proj, wpT, Cc, Cc);
  cvt_transpose_f16<<<dim3(HID / 32, Cc / 32), 256, 0, stream>>>(w_mlp1, w1T, Cc, HID);
  cvt_transpose_f16<<<dim3(Cc / 32, HID / 32), 256, 0, stream>>>(w_mlp2, w2T, HID, Cc);

  // LN1 -> xn (f16)
  layernorm_rows<<<Rows, 256, 0, stream>>>(x, ln1_g, ln1_b, xn, Cc);
  // QKV GEMM: [4096,1024] x [1024,3072] -> f16
  gemm_wmma<0><<<dim3(3 * Cc / BN, Rows / BM), 256, 0, stream>>>(
      xn, wqT, Rows, 3 * Cc, Cc, nullptr, nullptr, qkvb, nullptr);
  // V -> V^T per (b,h)
  transpose_v<<<dim3(Nn / 32, 2, Bb * Hh), 256, 0, stream>>>(qkvb, vTb, Nn);
  // flash attention -> yatt (f16)
  attn_flash<<<dim3(Nn / 128, Hh, Bb), 256, 0, stream>>>(qkvb, vTb, yatt, Nn);
  // proj + bias + residual(x) -> x1 (f32)
  gemm_wmma<1><<<dim3(Cc / BN, Rows / BM), 256, 0, stream>>>(
      yatt, wpT, Rows, Cc, Cc, b_proj, x, nullptr, x1);
  // LN2 -> xn (f16)
  layernorm_rows<<<Rows, 256, 0, stream>>>(x1, ln2_g, ln2_b, xn, Cc);
  // MLP1 + bias + GELU -> hbuf (f16)
  gemm_wmma<2><<<dim3(HID / BN, Rows / BM), 256, 0, stream>>>(
      xn, w1T, Rows, HID, Cc, b_mlp1, nullptr, hbuf, nullptr);
  // MLP2 + bias + residual(x1) -> out (f32)
  gemm_wmma<1><<<dim3(Cc / BN, Rows / BM), 256, 0, stream>>>(
      hbuf, w2T, Rows, Cc, HID, b_mlp2, x1, nullptr, out);
}
